// GraphEncoder_35699768164477
// MI455X (gfx1250) — compile-verified, matrix operand order checked by reading, not run on hardware
//
#include <hip/hip_runtime.h>
#include <hip/hip_bf16.h>

typedef __attribute__((ext_vector_type(16))) _Float16 v16h;
typedef __attribute__((ext_vector_type(8)))  float    v8f;

#define NNODES 25000
#define KADJ   16
#define HDIM   128
#define EDIM   128
#define BQ     50
#define LQ     128
#define NTOK   4

__device__ __forceinline__ float sigmf(float x) { return 1.0f / (1.0f + __expf(-x)); }

// ---------------------------------------------------------------------------
// Weight pre-pack: f32 -> f16, fragment-major layout
//   out[((nt*(Kd/32) + kc)*32 + lane)*16 + e]
// lane lo=lane&15, hi=lane>>4 ; n = nt*16+lo ; k = kc*32 + 8*hi + (e<8 ? e : 8+e)
// wT==1: W is (Nd x Kd) row-major.  wT==0: W is (Kd x Nd) row-major.
// ---------------------------------------------------------------------------
__global__ void pack_w(const float* __restrict__ W, _Float16* __restrict__ out,
                       int Nd, int Kd, int wT)
{
    const int i = blockIdx.x * 256 + threadIdx.x;
    if (i >= Nd * Kd) return;
    const int e    = i & 15;
    const int lane = (i >> 4) & 31;
    const int rest = i >> 9;               // nt*(Kd/32) + kc
    const int kt   = Kd >> 5;
    const int kc   = rest % kt;
    const int nt   = rest / kt;
    const int lo = lane & 15, hi = lane >> 4;
    const int n = nt * 16 + lo;
    const int k = kc * 32 + 8 * hi + (e < 8 ? e : 8 + e);
    const float v = wT ? W[(size_t)n * Kd + k] : W[(size_t)k * Nd + n];
    out[i] = (_Float16)v;
}

// ---------------------------------------------------------------------------
// WMMA GEMM: Out = act(A(MxK) * Wpacked + bias); one wave computes 16x64.
// ---------------------------------------------------------------------------
template <int ACT>
__global__ __launch_bounds__(32)
void wmma_gemm(const float* __restrict__ A, const _Float16* __restrict__ Wp,
               const float* __restrict__ bias, float* __restrict__ Out,
               int M, int Kd, int Nd, int ldo)
{
    const int lane = threadIdx.x & 31;
    const int lo = lane & 15, hi = lane >> 4;
    const int m0 = blockIdx.x * 16;
    const int nt0 = blockIdx.y * 4;        // first of 4 N-tiles
    const int kt = Kd >> 5;

    v8f acc[4] = {};
    int r = m0 + lo; if (r >= M) r = M - 1;
    const float* arow = A + (size_t)r * Kd;

#pragma unroll 1
    for (int kc = 0; kc < kt; ++kc) {
        const float4* pa = (const float4*)(arow + kc * 32 + 8 * hi);
        const float4 a0 = pa[0], a1 = pa[1], a2 = pa[4], a3 = pa[5];
        v16h af;
        af[0] = (_Float16)a0.x;  af[1] = (_Float16)a0.y;
        af[2] = (_Float16)a0.z;  af[3] = (_Float16)a0.w;
        af[4] = (_Float16)a1.x;  af[5] = (_Float16)a1.y;
        af[6] = (_Float16)a1.z;  af[7] = (_Float16)a1.w;
        af[8] = (_Float16)a2.x;  af[9] = (_Float16)a2.y;
        af[10] = (_Float16)a2.z; af[11] = (_Float16)a2.w;
        af[12] = (_Float16)a3.x; af[13] = (_Float16)a3.y;
        af[14] = (_Float16)a3.z; af[15] = (_Float16)a3.w;
#pragma unroll
        for (int tn = 0; tn < 4; ++tn) {
            const v16h bf = *(const v16h*)(Wp +
                (((size_t)(nt0 + tn) * kt + kc) * 32 + lane) * 16);
            acc[tn] = __builtin_amdgcn_wmma_f32_16x16x32_f16(
                false, af, false, bf, (short)0, acc[tn], false, false);
        }
    }

#pragma unroll
    for (int tn = 0; tn < 4; ++tn) {
        const int col = (nt0 + tn) * 16 + lo;
        const float bv = bias ? bias[col] : 0.0f;
#pragma unroll
        for (int v = 0; v < 8; ++v) {
            const int row = m0 + v + 8 * hi;
            if (row < M) {
                float x = acc[tn][v] + bv;
                if (ACT == 1)      x = fmaxf(x, 0.0f);
                else if (ACT == 2) x = sigmf(x);
                Out[(size_t)row * ldo + col] = x;
            }
        }
    }
}

// ---------------------------------------------------------------------------
// LSTM recurrence. One wave handles 16 rows for T steps.
// xg: [m*T + t]*512 (precomputed x@Wih^T + b). Whp: packed (512x128) fragments.
// ---------------------------------------------------------------------------
__global__ __launch_bounds__(32)
void lstm_rec(const float* __restrict__ xg, const _Float16* __restrict__ Whp,
              float* __restrict__ out_all, float* __restrict__ out_last,
              int M, int T)
{
    __shared__ float hL[16 * 128];
    __shared__ float cL[16 * 128];
    const int lane = threadIdx.x & 31;
    const int lo = lane & 15, hi = lane >> 4;
    const int m0 = blockIdx.x * 16;

    for (int i = lane; i < 16 * 128; i += 32) { hL[i] = 0.0f; cL[i] = 0.0f; }
    __syncthreads();

#pragma unroll 1
    for (int t = 0; t < T; ++t) {
        // Capture h fragments for this step (4 k-chunks of 32)
        v16h af[4];
#pragma unroll
        for (int kc = 0; kc < 4; ++kc) {
            const float4* ph = (const float4*)(hL + lo * 128 + kc * 32 + 8 * hi);
            const float4 a0 = ph[0], a1 = ph[1], a2 = ph[4], a3 = ph[5];
            af[kc][0] = (_Float16)a0.x;  af[kc][1] = (_Float16)a0.y;
            af[kc][2] = (_Float16)a0.z;  af[kc][3] = (_Float16)a0.w;
            af[kc][4] = (_Float16)a1.x;  af[kc][5] = (_Float16)a1.y;
            af[kc][6] = (_Float16)a1.z;  af[kc][7] = (_Float16)a1.w;
            af[kc][8] = (_Float16)a2.x;  af[kc][9] = (_Float16)a2.y;
            af[kc][10] = (_Float16)a2.z; af[kc][11] = (_Float16)a2.w;
            af[kc][12] = (_Float16)a3.x; af[kc][13] = (_Float16)a3.y;
            af[kc][14] = (_Float16)a3.z; af[kc][15] = (_Float16)a3.w;
        }
#pragma unroll 1
        for (int jt = 0; jt < 8; ++jt) {
            v8f acc[4] = {};
#pragma unroll
            for (int g = 0; g < 4; ++g) {          // gates i,f,g,o
                const int nt = g * 8 + jt;         // N-tile of Whh output dim
#pragma unroll
                for (int kc = 0; kc < 4; ++kc) {
                    const v16h bf = *(const v16h*)(Whp +
                        (((size_t)nt * 4 + kc) * 32 + lane) * 16);
                    acc[g] = __builtin_amdgcn_wmma_f32_16x16x32_f16(
                        false, af[kc], false, bf, (short)0, acc[g], false, false);
                }
            }
            const int col = jt * 16 + lo;
#pragma unroll
            for (int v = 0; v < 8; ++v) {
                const int row = v + 8 * hi;
                const int m = m0 + row;
                const int mc = (m < M) ? m : (M - 1);
                const float* xr = xg + ((size_t)mc * T + t) * 512;
                const float gi = acc[0][v] + xr[col];
                const float gf = acc[1][v] + xr[128 + col];
                const float gg = acc[2][v] + xr[256 + col];
                const float go = acc[3][v] + xr[384 + col];
                float cc = cL[row * 128 + col];
                cc = sigmf(gf) * cc + sigmf(gi) * tanhf(gg);
                const float hh = sigmf(go) * tanhf(cc);
                cL[row * 128 + col] = cc;
                hL[row * 128 + col] = hh;
                if (m < M) {
                    if (out_all)                out_all[((size_t)m * T + t) * 128 + col] = hh;
                    if (out_last && t == T - 1) out_last[(size_t)m * 128 + col] = hh;
                }
            }
        }
    }
}

// ---------------------------------------------------------------------------
// Gated attention context: one wave per node.
// ---------------------------------------------------------------------------
__global__ __launch_bounds__(32)
void attn_ctx(const float* __restrict__ hid, const int* __restrict__ adj,
              const float* __restrict__ att0, const float* __restrict__ att1,
              const float* __restrict__ gate, float* __restrict__ gctx,
              int Nn, int D)
{
    const int n = blockIdx.x;
    const int lane = threadIdx.x & 31;
    const float* self = hid + (size_t)n * D;

    int nbr[KADJ];
#pragma unroll
    for (int k = 0; k < KADJ; ++k) nbr[k] = adj[(size_t)n * KADJ + k];

    float s = 0.0f;
    for (int d = lane; d < D; d += 32) s += self[d] * att0[d];
#pragma unroll
    for (int off = 16; off; off >>= 1) s += __shfl_xor(s, off, 32);

    float sc[KADJ];
#pragma unroll 1
    for (int k = 0; k < KADJ; ++k) {
        float v = 0.0f;
        if (nbr[k] < Nn) {
            const float* p = hid + (size_t)nbr[k] * D;
            for (int d = lane; d < D; d += 32) v += p[d] * att1[d];
        }
#pragma unroll
        for (int off = 16; off; off >>= 1) v += __shfl_xor(v, off, 32);
        sc[k] = v + s;
    }
    float mx = sc[0];
#pragma unroll
    for (int k = 1; k < KADJ; ++k) mx = fmaxf(mx, sc[k]);
    float se = 0.0f;
#pragma unroll
    for (int k = 0; k < KADJ; ++k) { sc[k] = __expf(sc[k] - mx); se += sc[k]; }
    const float inv = 1.0f / se;

    for (int d = lane; d < D; d += 32) {
        float ctx = 0.0f;
#pragma unroll 1
        for (int k = 0; k < KADJ; ++k) {
            if (nbr[k] < Nn) ctx += sc[k] * hid[(size_t)nbr[k] * D + d];
        }
        gctx[(size_t)n * D + d] = gate[(size_t)n * D + d] * ctx * inv;
    }
}

// ---------------------------------------------------------------------------
// Streaming helpers
// ---------------------------------------------------------------------------
__global__ void gather_rows(const float* __restrict__ tab, const int* __restrict__ idx,
                            float* __restrict__ out, long rows, int E)
{
    const long i = (long)blockIdx.x * blockDim.x + threadIdx.x;
    if (i >= rows * E) return;
    const long r = i / E;
    const int  e = (int)(i - r * E);
    out[i] = tab[(size_t)idx[r] * E + e];
}

__global__ void build_rev(const float* __restrict__ se, const int* __restrict__ lens,
                          float* __restrict__ out, int Bq, int Lq, int E)
{
    const long i = (long)blockIdx.x * blockDim.x + threadIdx.x;
    if (i >= (long)Bq * Lq * E) return;
    const int b = (int)(i / ((long)Lq * E));
    const int rem = (int)(i - (long)b * Lq * E);
    const int t = rem / E, e = rem - t * E;
    int ri = lens[b] - 1 - t;
    ri = ri < 0 ? 0 : (ri > Lq - 1 ? Lq - 1 : ri);
    out[i] = se[((size_t)b * Lq + ri) * E + e];
}

__global__ void seq_combine(const float* __restrict__ hsf, const float* __restrict__ hsb,
                            const int* __restrict__ lens, float* __restrict__ enc,
                            int Bq, int Lq, int Hh)
{
    const long i = (long)blockIdx.x * blockDim.x + threadIdx.x;
    const int  D2 = 2 * Hh;
    if (i >= (long)Bq * Lq * D2) return;
    const int b = (int)(i / ((long)Lq * D2));
    const int rem = (int)(i - (long)b * Lq * D2);
    const int t = rem / D2, j = rem - t * D2;
    const int ln = lens[b];
    float v;
    if (j < Hh) {
        v = hsf[((size_t)b * Lq + t) * Hh + j];
    } else {
        int ri = ln - 1 - t;
        ri = ri < 0 ? 0 : (ri > Lq - 1 ? Lq - 1 : ri);
        v = hsb[((size_t)b * Lq + ri) * Hh + (j - Hh)];
    }
    enc[i] = (t < ln) ? v : 0.0f;
}

__global__ void make_mask(const int* __restrict__ idx, float* __restrict__ mask, int count)
{
    const int i = blockIdx.x * blockDim.x + threadIdx.x;
    if (i < count) mask[i] = (idx[i] == 0) ? 1.0f : 0.0f;
}

__global__ void final_hidden(const float* __restrict__ fw, const float* __restrict__ bw,
                             float* __restrict__ gh, int Nn, int Dh)
{
    const long i = (long)blockIdx.x * blockDim.x + threadIdx.x;
    if (i >= (long)Nn * 2 * Dh) return;
    const long n = i / (2 * Dh);
    const int  j = (int)(i - n * 2 * Dh);
    const float v = (j < Dh) ? fw[(size_t)n * Dh + j] : bw[(size_t)n * Dh + (j - Dh)];
    gh[i] = fmaxf(v, 0.0f);
}

__global__ void graph_max(const float* __restrict__ gh, float* __restrict__ ge,
                          int G, int Lg, int C)
{
    const int i = blockIdx.x * blockDim.x + threadIdx.x;
    if (i >= G * C) return;
    const int g = i / C, c = i - g * C;
    float m = -3.4e38f;
    for (int r = 0; r < Lg; ++r)
        m = fmaxf(m, gh[((size_t)g * Lg + r) * C + c]);
    ge[i] = m;
}

__global__ void write_maxlen(const int* __restrict__ ml, float* __restrict__ out)
{
    if (threadIdx.x == 0 && blockIdx.x == 0) out[0] = (float)ml[0];
}

// ---------------------------------------------------------------------------
extern "C" void kernel_launch(void* const* d_in, const int* in_sizes, int n_in,
                              void* d_out, int out_size, void* d_ws, size_t ws_size,
                              hipStream_t stream)
{
    (void)in_sizes; (void)n_in; (void)out_size; (void)ws_size;
    const int Nn = NNODES, Hh = HDIM, E = EDIM, B = BQ, L = LQ, NT = NTOK;

    const float* embed  = (const float*)d_in[0];
    const float* sf_Wih = (const float*)d_in[1];
    const float* sf_Whh = (const float*)d_in[2];
    const float* sf_b   = (const float*)d_in[3];
    const float* sb_Wih = (const float*)d_in[4];
    const float* sb_Whh = (const float*)d_in[5];
    const float* sb_b   = (const float*)d_in[6];
    const float* nd_Wih = (const float*)d_in[7];
    const float* nd_Whh = (const float*)d_in[8];
    const float* nd_b   = (const float*)d_in[9];
    const float* dir_att0[2] = { (const float*)d_in[10], (const float*)d_in[20] };
    const float* dir_attr[2] = { (const float*)d_in[11], (const float*)d_in[21] };
    const float* dir_Wg0[2]  = { (const float*)d_in[12], (const float*)d_in[22] };
    const float* dir_bg0[2]  = { (const float*)d_in[13], (const float*)d_in[23] };
    const float* dir_Wgr[2]  = { (const float*)d_in[14], (const float*)d_in[24] };
    const float* dir_bgr[2]  = { (const float*)d_in[15], (const float*)d_in[25] };
    const float* dir_Wo0[2]  = { (const float*)d_in[16], (const float*)d_in[26] };
    const float* dir_bo0[2]  = { (const float*)d_in[17], (const float*)d_in[27] };
    const float* dir_Wor[2]  = { (const float*)d_in[18], (const float*)d_in[28] };
    const float* dir_bor[2]  = { (const float*)d_in[19], (const float*)d_in[29] };
    const int* node_feat  = (const int*)d_in[30];
    const int* dir_adj[2] = { (const int*)d_in[31], (const int*)d_in[32] };
    const int* idx_seqs   = (const int*)d_in[33];
    const int* seq_lens   = (const int*)d_in[34];
    const int* max_len_p  = (const int*)d_in[35];

    // ---- output layout (flat float) ----
    float* out_f = (float*)d_out;
    float* gh   = out_f;                              // (25,1000,512)
    float* ge   = gh + (size_t)25 * 1000 * 512;       // (25,512)
    float* mlo  = ge + (size_t)25 * 512;              // (1,)
    float* enc  = mlo + 1;                            // (50,128,256)
    float* mask = enc + (size_t)B * L * 2 * Hh;       // (50,128)

    // ---- workspace layout (f32 part) ----
    float* w = (float*)d_ws;
    float* se      = w; w += (size_t)B * L * E;
    float* se_rev  = w; w += (size_t)B * L * E;
    float* xg_f    = w; w += (size_t)B * L * 4 * Hh;
    float* xg_b    = w; w += (size_t)B * L * 4 * Hh;
    float* hs_f    = w; w += (size_t)B * L * Hh;
    float* hs_b    = w; w += (size_t)B * L * Hh;
    float* ntok    = w; w += (size_t)Nn * NT * E;
    float* nxg     = w; w += (size_t)Nn * NT * 4 * Hh;
    float* nemb    = w; w += (size_t)Nn * Hh;
    float* hbuf[2][2];
    for (int d = 0; d < 2; ++d)
        for (int p = 0; p < 2; ++p) { hbuf[d][p] = w; w += (size_t)Nn * 256; }
    float* gatebuf = w; w += (size_t)Nn * 256;
    float* gctxbuf = w; w += (size_t)Nn * 256;

    // ---- packed f16 weights ----
    _Float16* hw = (_Float16*)w;
    auto alloc_h = [&](size_t n) { _Float16* p = hw; hw += n; return p; };
    _Float16* pWih_f = alloc_h((size_t)512 * 128);
    _Float16* pWih_b = alloc_h((size_t)512 * 128);
    _Float16* pWih_n = alloc_h((size_t)512 * 128);
    _Float16* pWhh_f = alloc_h((size_t)512 * 128);
    _Float16* pWhh_b = alloc_h((size_t)512 * 128);
    _Float16* pWhh_n = alloc_h((size_t)512 * 128);
    _Float16 *pWg[2][3], *pWo0[2][3], *pWo1[2][3];
    for (int d = 0; d < 2; ++d)
        for (int l = 0; l < 3; ++l) {
            const int Din = (l == 0) ? Hh : 2 * Hh;
            pWg[d][l]  = alloc_h((size_t)Din * Din);
            pWo0[d][l] = alloc_h((size_t)Din * Hh);
            pWo1[d][l] = alloc_h((size_t)Din * Hh);
        }

    auto pack = [&](const float* W, _Float16* out, int Nd, int Kd, int wT) {
        const int cnt = Nd * Kd;
        pack_w<<<(cnt + 255) / 256, 256, 0, stream>>>(W, out, Nd, Kd, wT);
    };
    auto gemm = [&](const float* A, const _Float16* Wp, const float* bias, float* Out,
                    int M, int Kd, int Nd, int ldo, int act) {
        dim3 grid((M + 15) / 16, Nd / 64);
        if (act == 0)      wmma_gemm<0><<<grid, 32, 0, stream>>>(A, Wp, bias, Out, M, Kd, Nd, ldo);
        else if (act == 1) wmma_gemm<1><<<grid, 32, 0, stream>>>(A, Wp, bias, Out, M, Kd, Nd, ldo);
        else               wmma_gemm<2><<<grid, 32, 0, stream>>>(A, Wp, bias, Out, M, Kd, Nd, ldo);
    };

    // ---- pack all weights once ----
    pack(sf_Wih, pWih_f, 512, 128, 1);
    pack(sb_Wih, pWih_b, 512, 128, 1);
    pack(nd_Wih, pWih_n, 512, 128, 1);
    pack(sf_Whh, pWhh_f, 512, 128, 1);
    pack(sb_Whh, pWhh_b, 512, 128, 1);
    pack(nd_Whh, pWhh_n, 512, 128, 1);
    for (int d = 0; d < 2; ++d)
        for (int l = 0; l < 3; ++l) {
            const int Din = (l == 0) ? Hh : 2 * Hh;
            const float *Wg, *Wo0p, *Wo1p;
            if (l == 0) {
                Wg   = dir_Wg0[d];
                Wo0p = dir_Wo0[d];
                Wo1p = dir_Wo0[d] + Hh * Hh;
            } else {
                const int j = l - 1;
                Wg   = dir_Wgr[d] + (size_t)j * 256 * 256;
                Wo0p = dir_Wor[d] + (size_t)(j * 2 + 0) * 256 * Hh;
                Wo1p = dir_Wor[d] + (size_t)(j * 2 + 1) * 256 * Hh;
            }
            pack(Wg,   pWg[d][l],  Din, Din, 0);
            pack(Wo0p, pWo0[d][l], Hh,  Din, 0);
            pack(Wo1p, pWo1[d][l], Hh,  Din, 0);
        }

    // ================= sequence encoder =================
    {
        long cnt = (long)B * L * E;
        gather_rows<<<(cnt + 255) / 256, 256, 0, stream>>>(embed, idx_seqs, se, (long)B * L, E);
        build_rev<<<(cnt + 255) / 256, 256, 0, stream>>>(se, seq_lens, se_rev, B, L, E);
    }
    gemm(se,     pWih_f, sf_b, xg_f, B * L, E, 4 * Hh, 4 * Hh, 0);
    gemm(se_rev, pWih_b, sb_b, xg_b, B * L, E, 4 * Hh, 4 * Hh, 0);
    lstm_rec<<<(B + 15) / 16, 32, 0, stream>>>(xg_f, pWhh_f, hs_f, nullptr, B, L);
    lstm_rec<<<(B + 15) / 16, 32, 0, stream>>>(xg_b, pWhh_b, hs_b, nullptr, B, L);
    {
        long cnt = (long)B * L * 2 * Hh;
        seq_combine<<<(cnt + 255) / 256, 256, 0, stream>>>(hs_f, hs_b, seq_lens, enc, B, L, Hh);
        make_mask<<<(B * L + 255) / 256, 256, 0, stream>>>(idx_seqs, mask, B * L);
    }

    // ================= node embedding (LSTM over NT tokens) =================
    {
        long cnt = (long)Nn * NT * E;
        gather_rows<<<(cnt + 255) / 256, 256, 0, stream>>>(embed, node_feat, ntok, (long)Nn * NT, E);
    }
    gemm(ntok, pWih_n, nd_b, nxg, Nn * NT, E, 4 * Hh, 4 * Hh, 0);
    lstm_rec<<<(Nn + 15) / 16, 32, 0, stream>>>(nxg, pWhh_n, nullptr, nemb, Nn, NT);

    // ================= GGNN layers (2 directions x 3 layers) =================
    for (int d = 0; d < 2; ++d) {
        for (int l = 0; l < 3; ++l) {
            const int Din = (l == 0) ? Hh : 2 * Hh;
            const float* hin  = (l == 0) ? nemb : hbuf[d][(l + 1) & 1];
            float*       hout = hbuf[d][l & 1];
            const float *bg, *a0, *a1, *bo0p, *bo1p;
            if (l == 0) {
                bg = dir_bg0[d];
                a0 = dir_att0[d]; a1 = dir_att0[d] + Hh;
                bo0p = dir_bo0[d]; bo1p = dir_bo0[d] + Hh;
            } else {
                const int j = l - 1;
                bg = dir_bgr[d] + (size_t)j * 256;
                a0 = dir_attr[d] + (size_t)j * 2 * 256; a1 = a0 + 256;
                bo0p = dir_bor[d] + (size_t)(j * 2 + 0) * Hh;
                bo1p = dir_bor[d] + (size_t)(j * 2 + 1) * Hh;
            }
            // gate = sigmoid(hin @ Wg + bg)
            gemm(hin, pWg[d][l], bg, gatebuf, Nn, Din, Din, Din, 2);
            // gctx = gate * softmax-ctx(neighbors)
            attn_ctx<<<Nn, 32, 0, stream>>>(hin, dir_adj[d], a0, a1, gatebuf, gctxbuf, Nn, Din);
            // hout[:, :128]  = relu(hin  @ Wo0 + bo0)
            gemm(hin,     pWo0[d][l], bo0p, hout,      Nn, Din, Hh, 256, 1);
            // hout[:, 128:] = relu(gctx @ Wo1 + bo1)
            gemm(gctxbuf, pWo1[d][l], bo1p, hout + Hh, Nn, Din, Hh, 256, 1);
        }
    }

    // ================= epilogue =================
    {
        long cnt = (long)Nn * 512;
        final_hidden<<<(cnt + 255) / 256, 256, 0, stream>>>(hbuf[0][0], hbuf[1][0], gh, Nn, 256);
        graph_max<<<(25 * 512 + 255) / 256, 256, 0, stream>>>(gh, ge, 25, 1000, 512);
        write_maxlen<<<1, 32, 0, stream>>>(max_len_p, mlo);
    }
}
